// AttenPool_22917945491863
// MI455X (gfx1250) — compile-verified, matrix operand order checked by reading, not run on hardware
//
#include <hip/hip_runtime.h>
#include <hip/hip_bf16.h>

// AttenPool, MI455X (gfx1250), wave32.
// Identity: sum_L softmax(axis=L) == 1 => attention mask == 1
// => out = 4x4 sum-pool of ((1-alpha)*(conv3(bn(x),wv)+bv) + alpha*x).
// Only the value conv survives: 19.3 GFLOP via v_wmma_f32_16x16x32_f16.
// Halo staging uses GLOBAL_LOAD_ASYNC_TO_LDS_B128 (ASYNCcnt path);
// weight A-fragments are software-pipelined to hide global-load latency.

typedef __attribute__((ext_vector_type(16))) _Float16 v16h;
typedef __attribute__((ext_vector_type(8)))  float    v8f;

#define B_  16
#define C_  64
#define H_  128
#define W_  128

// ---------------------------------------------------------------------------
// Kernel 1: fold BatchNorm into x, store f16 NHWC: xh[b][h][w][ci]
// ---------------------------------------------------------------------------
__global__ __launch_bounds__(256)
void prep_bn_kernel(const float* __restrict__ x,
                    const float* __restrict__ g0, const float* __restrict__ b0,
                    const float* __restrict__ m0, const float* __restrict__ v0,
                    _Float16* __restrict__ xh) {
    int pix = blockIdx.x * 256 + threadIdx.x;     // 16*128*128 = 262144 pixels
    int b = pix >> 14;
    int hw = pix & 16383;
    int h = hw >> 7;
    int w = hw & 127;

    _Float16 tmp[C_];
#pragma unroll
    for (int ci = 0; ci < C_; ++ci) {
        float scale = g0[ci] * rsqrtf(v0[ci] + 1e-5f);
        float val = (x[(((b * C_ + ci) * H_) + h) * W_ + w] - m0[ci]) * scale + b0[ci];
        tmp[ci] = (_Float16)val;
    }
    uint4* dst = (uint4*)(xh + (size_t)pix * C_);
    const uint4* src = (const uint4*)tmp;
#pragma unroll
    for (int i = 0; i < 8; ++i) dst[i] = src[i];
}

// ---------------------------------------------------------------------------
// Kernel 2: pre-swizzle conv weights into WMMA A-fragment layout.
// wA[s][mt][lane][e] with s = t*2+kt; one contiguous 32B load per fragment.
// A layout (ISA 7.12.2, 16-bit 16x32): M = lane%16,
//   K = (e/8)*16 + (lane>=16 ? 8 : 0) + (e%8), within K-tile kt of 32.
// ---------------------------------------------------------------------------
__global__ __launch_bounds__(256)
void pack_w_kernel(const float* __restrict__ wv, _Float16* __restrict__ wA) {
    int idx = blockIdx.x * 256 + threadIdx.x;     // 9*2*4*32*16 = 36864
    if (idx >= 9 * 2 * 4 * 32 * 16) return;
    int e    = idx & 15;
    int lane = (idx >> 4) & 31;
    int mt   = (idx >> 9) & 3;
    int kt   = (idx >> 11) & 1;
    int t    = idx >> 12;                         // 0..8 tap
    int co   = mt * 16 + (lane & 15);
    int grp  = e >> 3;
    int off  = e & 7;
    int hi8  = (lane >= 16) ? 8 : 0;
    int ci   = kt * 32 + grp * 16 + hi8 + off;
    int kh   = t / 3, kw = t % 3;
    wA[idx] = (_Float16)wv[((co * C_ + ci) * 3 + kh) * 3 + kw];
}

// ---------------------------------------------------------------------------
// Kernel 3: implicit-GEMM conv (9 taps x 2 K-steps) + bias + alpha-residual
// + 4x4 sum-pool. Block = 128 threads = 4 waves; wave mt owns output channels
// [16mt, 16mt+16). Block tile = 4 rows x 16 cols (one patch row); halo tile
// [6 rows][18 cols][64 ci] f16 staged straight into LDS with async loads.
// ---------------------------------------------------------------------------
__global__ __launch_bounds__(128)
void conv_pool_kernel(const _Float16* __restrict__ xh,
                      const _Float16* __restrict__ wA,
                      const float* __restrict__ x,
                      const float* __restrict__ bv,
                      const float* __restrict__ alphaP,
                      float* __restrict__ out) {
    __shared__ _Float16 ldsX[6 * 18 * C_];        // 13824 bytes

    const int tid  = threadIdx.x;
    const int mt   = tid >> 5;                    // wave id: M-tile
    const int lane = tid & 31;
    const int lcol = lane & 15;
    const int hi8  = (lane >= 16) ? 8 : 0;

    const int bid = blockIdx.x;                   // 16 * 32 * 8 = 4096
    const int b   = bid >> 8;
    const int ph  = (bid >> 3) & 31;              // patch row
    const int wb  = bid & 7;                      // 16-wide column block
    const int h0  = ph << 2;
    const int w0  = wb << 4;

    // ---- stage halo tile: rows h0-1..h0+4, cols w0-1..w0+16, all 64 ci ----
    // Direct global->LDS async copy (no VGPR round trip), ASYNCcnt-tracked.
    for (int chunk = tid; chunk < 108 * 8; chunk += 128) {
        int rc   = chunk >> 3;                    // (row,col) pair, 0..107
        int part = chunk & 7;                     // 16B sub-chunk of 64 ci
        int row  = rc / 18;
        int col  = rc % 18;
        int hr   = h0 - 1 + row;
        int wc   = w0 - 1 + col;
        _Float16* lp = &ldsX[(row * 18 + col) * C_ + part * 8];
        if (hr >= 0 && hr < H_ && wc >= 0 && wc < W_) {
            const _Float16* gp =
                xh + ((size_t)((b * H_ + hr) * W_ + wc) * C_ + part * 8);
            unsigned ldsoff = (unsigned)(size_t)lp;   // low 32 bits = LDS byte addr
            asm volatile("global_load_async_to_lds_b128 %0, %1, off"
                         :: "v"(ldsoff), "v"(gp) : "memory");
        } else {
            *(uint4*)lp = make_uint4(0u, 0u, 0u, 0u);
        }
    }
    asm volatile("s_wait_asynccnt 0x0" ::: "memory");
    __syncthreads();

    // ---- 9-tap implicit GEMM: D[co, pos] += Wtap[co,ci] * xn[ci, pos+tap] ----
    v8f c[4] = {v8f{}, v8f{}, v8f{}, v8f{}};

    // software-pipelined A-fragment loads: fetch K-step s+1 during K-step s
    v16h afrag = *(const v16h*)(wA + (((0 * 4 + mt) * 32 + lane) << 4));
#pragma unroll
    for (int s = 0; s < 18; ++s) {                // s = t*2 + kt
        v16h anext = afrag;
        if (s < 17) {
            anext = *(const v16h*)(wA + ((((s + 1) * 4 + mt) * 32 + lane) << 4));
        }
        const int t  = s >> 1;
        const int kt = s & 1;
        const int kh = t / 3, kw = t % 3;
        const int cibase = kt * 32 + hi8;
#pragma unroll
        for (int nt = 0; nt < 4; ++nt) {          // N-tile = image row h0+nt
            union { v16h v; uint4 q[2]; } bu;
            const _Float16* p = &ldsX[((nt + kh) * 18 + (lcol + kw)) * C_ + cibase];
            bu.q[0] = *(const uint4*)(p);          // K offsets  0..7  (pair-group 0)
            bu.q[1] = *(const uint4*)(p + 16);     // K offsets 16..23 (pair-group 1)
            c[nt] = __builtin_amdgcn_wmma_f32_16x16x32_f16(
                false, afrag, false, bu.v, (short)0, c[nt], false, false);
        }
        afrag = anext;
    }

    // ---- epilogue: bias + alpha-residual + 4-row & 4-lane patch reduction ----
    const float alpha = alphaP[0];
    const float oma   = 1.0f - alpha;

#pragma unroll
    for (int r = 0; r < 8; ++r) {
        const int co   = mt * 16 + r + hi8;       // C/D layout: VGPR r -> M=r / r+8
        const float bias = bv[co];
        float s = 0.0f;
#pragma unroll
        for (int nt = 0; nt < 4; ++nt) {
            float xv = x[((b * C_ + co) * H_ + (h0 + nt)) * W_ + w0 + lcol];
            s += oma * (c[nt][r] + bias) + alpha * xv;
        }
        // sum over 4 adjacent columns (one 4-wide patch) within the wave
        s += __shfl_xor(s, 1);
        s += __shfl_xor(s, 2);
        if ((lane & 3) == 0) {
            int ow = (w0 >> 2) + (lcol >> 2);
            out[((b * C_ + co) * 32 + ph) * 32 + ow] = s;
        }
    }
}

// ---------------------------------------------------------------------------
extern "C" void kernel_launch(void* const* d_in, const int* in_sizes, int n_in,
                              void* d_out, int out_size, void* d_ws, size_t ws_size,
                              hipStream_t stream) {
    (void)in_sizes; (void)n_in; (void)out_size; (void)ws_size;

    const float* x     = (const float*)d_in[0];
    const float* g0    = (const float*)d_in[1];
    const float* b0    = (const float*)d_in[2];
    const float* m0    = (const float*)d_in[3];
    const float* v0    = (const float*)d_in[4];
    // d_in[5..14]: query-branch params — mathematically eliminated (softmax-sum == 1)
    const float* wv    = (const float*)d_in[15];
    const float* bv    = (const float*)d_in[16];
    const float* alpha = (const float*)d_in[17];
    float* out = (float*)d_out;

    // workspace: xh (f16 NHWC, 33554432 B) | wA (packed weights, 73728 B)
    _Float16* xh = (_Float16*)d_ws;
    _Float16* wA = (_Float16*)((char*)d_ws + (size_t)B_ * H_ * W_ * C_ * sizeof(_Float16));

    prep_bn_kernel<<<(B_ * H_ * W_) / 256, 256, 0, stream>>>(x, g0, b0, m0, v0, xh);
    pack_w_kernel<<<144, 256, 0, stream>>>(wv, wA);
    conv_pool_kernel<<<B_ * 32 * 8, 128, 0, stream>>>(xh, wA, x, bv, alpha, out);
}
